// AttentionResidualGRU_13168369729790
// MI455X (gfx1250) — matrix operand order repaired and checked by reading, not compiled, not run on hardware
//
#include <hip/hip_runtime.h>

#define Bsz   256
#define Tsz   1024
#define Hsz   512
#define OUTsz 2
#define NBLK  128

typedef __bf16 bf16;
typedef bf16         v16bf  __attribute__((ext_vector_type(16)));
typedef float        v8f    __attribute__((ext_vector_type(8)));
typedef unsigned int uint4v __attribute__((ext_vector_type(4)));

union FragAB { v16bf v; uint4v q[2]; };

__device__ __forceinline__ float sigmoidf_(float x) {
    return 1.0f / (1.0f + __expf(-x));
}

// ---------------------------------------------------------------------------
// Init: zero both parities of the bf16 hidden-state mirror, zero the barrier
// counter. Runs every launch (harness requires determinism).
// ---------------------------------------------------------------------------
__global__ void init_kernel(bf16* __restrict__ hb, unsigned* __restrict__ cnt) {
    int i = blockIdx.x * blockDim.x + threadIdx.x;
    const int n = 2 * Bsz * Hsz;
    for (int idx = i; idx < n; idx += gridDim.x * blockDim.x)
        hb[idx] = (bf16)0.0f;
    if (i == 0) *cnt = 0u;
}

// ---------------------------------------------------------------------------
// Persistent GRU kernel: 128 blocks x 128 threads (4 waves).
// Block (rb, c): rows [rb*64, rb*64+64), H-columns [c*16, c*16+16) across all
// three gate sections of the 3H output. Wave w owns M-tile rb*4+w.
// W_hh slice lives in LDS; carried h lives in registers (WMMA C layout);
// bf16 h mirror is double-buffered in global for cross-block A-fragments.
// ---------------------------------------------------------------------------
__global__ __launch_bounds__(128) void gru_kernel(
    const float* __restrict__ V,     // [B, T, 2]
    const float* __restrict__ W_ih,  // [3H, 2]
    const float* __restrict__ W_hh,  // [3H, H]
    const float* __restrict__ b_ih,  // [3H]
    const float* __restrict__ b_hh,  // [3H]
    const float* __restrict__ Wr,    // [2, H]
    const float* __restrict__ br,    // [2]
    bf16*  __restrict__ hb,          // [2][B][H] double-buffered bf16 h
    float* __restrict__ res,         // [T][B][2]
    unsigned* __restrict__ cnt)
{
    __shared__ __align__(16) bf16 wlds[3][16][Hsz];   // 48 KB W_hh slice
    __shared__ float wi0s[3][16], wi1s[3][16], bihs[3][16], bhhs[3][16];
    __shared__ float vt[64][2];

    const int tid  = threadIdx.x;
    const int lane = tid & 31;
    const int w    = tid >> 5;
    const int c    = blockIdx.x & 31;   // column tile index within H
    const int rb   = blockIdx.x >> 5;   // row group (64 rows)
    const int m0   = rb * 64;
    const int mr   = m0 + w * 16;       // wave's M-tile base row
    const int half = lane >> 4;
    const int j    = lane & 15;         // within-tile column (C layout N)
    const int cg   = c * 16 + j;        // global H column

    // Stage W_hh slice (3 gates x 16 cols x 512 k), converting f32 -> bf16.
    for (int idx = tid; idx < 3 * 16 * Hsz; idx += 128) {
        int g   = idx >> 13;        // /8192
        int rem = idx & 8191;
        int n   = rem >> 9;         // /512
        int k   = rem & 511;
        wlds[g][n][k] = (bf16)W_hh[(g * Hsz + c * 16 + n) * Hsz + k];
    }
    if (tid < 48) {
        int g = tid >> 4, jj = tid & 15;
        int col = g * Hsz + c * 16 + jj;
        wi0s[g][jj] = W_ih[col * 2 + 0];
        wi1s[g][jj] = W_ih[col * 2 + 1];
        bihs[g][jj] = b_ih[col];
        bhhs[g][jj] = b_hh[col];
    }
    __syncthreads();

    float hreg[8];                       // carried fp32 h (WMMA C layout)
    #pragma unroll
    for (int i = 0; i < 8; ++i) hreg[i] = 0.0f;

    const float br0 = br[0], br1 = br[1];

    for (int t = 0; t < Tsz; ++t) {
        const bf16* hbR = hb + (size_t)(t & 1) * (Bsz * Hsz);       // h_t
        bf16*       hbW = hb + (size_t)((t + 1) & 1) * (Bsz * Hsz); // h_{t+1}

        // res_{t-1} = Wr . h_t + br, computed by c==0 blocks (deterministic).
        if (c == 0 && t > 0) {
            int rl = tid >> 1, o = tid & 1;
            const bf16*  hrow = hbR + (m0 + rl) * Hsz;
            const float* wrr  = Wr + o * Hsz;
            float s = 0.0f;
            for (int k = 0; k < Hsz; ++k) s += (float)hrow[k] * wrr[k];
            res[((t - 1) * Bsz + (m0 + rl)) * OUTsz + o] = s + (o ? br1 : br0);
        }

        // Stage V[:, t, :] for our 64 rows.
        {
            int r = tid >> 1, comp = tid & 1;
            vt[r][comp] = V[((m0 + r) * Tsz + t) * 2 + comp];
        }
        __syncthreads();

        // C[16x16] per gate: K-loop over H in steps of 32 (bf16 WMMA).
        v8f accR = {}, accZ = {}, accN = {};
        const bf16* hA = hbR + (mr + (lane & 15)) * Hsz;  // A row for this lane
        for (int kb = 0; kb < Hsz; kb += 32) {
            FragAB a;
            a.q[0] = *reinterpret_cast<const uint4v*>(hA + kb + half * 8);
            a.q[1] = *reinterpret_cast<const uint4v*>(hA + kb + 16 + half * 8);
            FragAB b0, b1, b2;
            const uint4v* p0 = reinterpret_cast<const uint4v*>(&wlds[0][j][kb + half * 8]);
            b0.q[0] = p0[0]; b0.q[1] = p0[2];
            const uint4v* p1 = reinterpret_cast<const uint4v*>(&wlds[1][j][kb + half * 8]);
            b1.q[0] = p1[0]; b1.q[1] = p1[2];
            const uint4v* p2 = reinterpret_cast<const uint4v*>(&wlds[2][j][kb + half * 8]);
            b2.q[0] = p2[0]; b2.q[1] = p2[2];
            accR = __builtin_amdgcn_wmma_f32_16x16x32_bf16(false, a.v, false, b0.v,
                                                           (short)0, accR, false, false);
            accZ = __builtin_amdgcn_wmma_f32_16x16x32_bf16(false, a.v, false, b1.v,
                                                           (short)0, accZ, false, false);
            accN = __builtin_amdgcn_wmma_f32_16x16x32_bf16(false, a.v, false, b2.v,
                                                           (short)0, accN, false, false);
        }

        // Gate combine + h update; publish bf16 h_{t+1}.
        #pragma unroll
        for (int i = 0; i < 8; ++i) {
            int rl   = w * 16 + i + (half << 3);   // block-local row (C layout M)
            int brow = m0 + rl;
            float v0 = vt[rl][0], v1 = vt[rl][1];
            float gxr = v0 * wi0s[0][j] + v1 * wi1s[0][j] + bihs[0][j];
            float gxz = v0 * wi0s[1][j] + v1 * wi1s[1][j] + bihs[1][j];
            float gxn = v0 * wi0s[2][j] + v1 * wi1s[2][j] + bihs[2][j];
            float rr = sigmoidf_(gxr + accR[i] + bhhs[0][j]);
            float zz = sigmoidf_(gxz + accZ[i] + bhhs[1][j]);
            float nn = tanhf(gxn + rr * (accN[i] + bhhs[2][j]));
            float hnew = (1.0f - zz) * nn + zz * hreg[i];
            hreg[i] = hnew;
            hbW[brow * Hsz + cg] = (bf16)hnew;
        }

        // Grid-wide step barrier (monotonic counter, device scope).
        __threadfence();
        __syncthreads();
        if (tid == 0) {
            __hip_atomic_fetch_add(cnt, 1u, __ATOMIC_RELEASE, __HIP_MEMORY_SCOPE_AGENT);
            unsigned tgt = (unsigned)NBLK * (unsigned)(t + 1);
            while (__hip_atomic_load(cnt, __ATOMIC_ACQUIRE, __HIP_MEMORY_SCOPE_AGENT) < tgt)
                __builtin_amdgcn_s_sleep(2);
        }
        __syncthreads();
        __threadfence();
    }

    // Epilogue: res_{T-1} from h_T (parity (T & 1) == 0).
    if (c == 0) {
        const bf16* hbR = hb + (size_t)(Tsz & 1) * (Bsz * Hsz);
        int rl = tid >> 1, o = tid & 1;
        const bf16*  hrow = hbR + (m0 + rl) * Hsz;
        const float* wrr  = Wr + o * Hsz;
        float s = 0.0f;
        for (int k = 0; k < Hsz; ++k) s += (float)hrow[k] * wrr[k];
        res[((Tsz - 1) * Bsz + (m0 + rl)) * OUTsz + o] = s + (o ? br1 : br0);
    }
}

// ---------------------------------------------------------------------------
// Correction scan: 256 blocks (one batch row each), 1 wave per block.
// Sequential in t, fully register/LDS-resident, shfl reductions.
// ---------------------------------------------------------------------------
__global__ __launch_bounds__(32) void corr_kernel(
    const float* __restrict__ X0,   // [B, 2]
    const float* __restrict__ V,    // [B, T, 2]
    const float* __restrict__ Wa1,  // [256, 4]
    const float* __restrict__ ba1,  // [256]
    const float* __restrict__ Wa2,  // [2, 256]
    const float* __restrict__ ba2,  // [2]
    const float* __restrict__ res,  // [T][B][2]
    float* __restrict__ out)        // [B, T, 2]
{
    __shared__ float lwa1[1024];
    __shared__ float lba1[256];
    __shared__ float lwa2[512];
    const int b    = blockIdx.x;
    const int lane = threadIdx.x;
    for (int i = lane; i < 1024; i += 32) lwa1[i] = Wa1[i];
    for (int i = lane; i < 256;  i += 32) lba1[i] = ba1[i];
    for (int i = lane; i < 512;  i += 32) lwa2[i] = Wa2[i];
    __syncthreads();

    float x0 = X0[b * 2 + 0], x1 = X0[b * 2 + 1];
    const float ba2_0 = ba2[0], ba2_1 = ba2[1];

    for (int t = 0; t < Tsz; ++t) {
        float v0 = V[(b * Tsz + t) * 2 + 0];
        float v1 = V[(b * Tsz + t) * 2 + 1];
        float r0 = res[(t * Bsz + b) * 2 + 0];
        float r1 = res[(t * Bsz + b) * 2 + 1];
        float s0 = 0.0f, s1 = 0.0f;
        #pragma unroll
        for (int ii = 0; ii < 8; ++ii) {
            int i = lane + ii * 32;
            float h = lwa1[4 * i + 0] * x0 + lwa1[4 * i + 1] * x1 +
                      lwa1[4 * i + 2] * v0 + lwa1[4 * i + 3] * v1 + lba1[i];
            h = fmaxf(h, 0.0f);
            s0 += lwa2[i] * h;
            s1 += lwa2[256 + i] * h;
        }
        #pragma unroll
        for (int m = 16; m >= 1; m >>= 1) {
            s0 += __shfl_xor(s0, m, 32);
            s1 += __shfl_xor(s1, m, 32);
        }
        float l0 = s0 + ba2_0, l1 = s1 + ba2_1;
        float mx = fmaxf(l0, l1);
        float e0 = __expf(l0 - mx), e1 = __expf(l1 - mx);
        float inv = 1.0f / (e0 + e1);
        x0 = x0 + v0 + (e0 * inv) * r0;
        x1 = x1 + v1 + (e1 * inv) * r1;
        if (lane == 0) {
            out[(b * Tsz + t) * 2 + 0] = x0;
            out[(b * Tsz + t) * 2 + 1] = x1;
        }
    }
}

// ---------------------------------------------------------------------------
extern "C" void kernel_launch(void* const* d_in, const int* in_sizes, int n_in,
                              void* d_out, int out_size, void* d_ws, size_t ws_size,
                              hipStream_t stream) {
    (void)in_sizes; (void)n_in; (void)out_size; (void)ws_size;
    const float* X0   = (const float*)d_in[0];
    const float* V    = (const float*)d_in[1];
    const float* W_ih = (const float*)d_in[2];
    const float* W_hh = (const float*)d_in[3];
    const float* b_ih = (const float*)d_in[4];
    const float* b_hh = (const float*)d_in[5];
    const float* Wa1  = (const float*)d_in[6];
    const float* ba1  = (const float*)d_in[7];
    const float* Wa2  = (const float*)d_in[8];
    const float* ba2  = (const float*)d_in[9];
    const float* Wr   = (const float*)d_in[10];
    const float* br   = (const float*)d_in[11];
    float* out = (float*)d_out;

    // Workspace layout: [counter pad 256B][hb: 2*256*512 bf16 = 512KB][res: 2MB]
    char* ws = (char*)d_ws;
    unsigned* cnt = (unsigned*)ws;
    bf16*  hb  = (bf16*)(ws + 256);
    float* res = (float*)(ws + 256 + 2 * Bsz * Hsz * sizeof(bf16));

    hipLaunchKernelGGL(init_kernel, dim3(512), dim3(256), 0, stream, hb, cnt);
    hipLaunchKernelGGL(gru_kernel, dim3(NBLK), dim3(128), 0, stream,
                       V, W_ih, W_hh, b_ih, b_hh, Wr, br, hb, res, cnt);
    hipLaunchKernelGGL(corr_kernel, dim3(Bsz), dim3(32), 0, stream,
                       X0, V, Wa1, ba1, Wa2, ba2, res, out);
}
